// Attention_26877905338569
// MI455X (gfx1250) — matrix operand order repaired and checked by reading, not compiled, not run on hardware
//
#include <hip/hip_runtime.h>
#include <hip/hip_bf16.h>

typedef __attribute__((ext_vector_type(16))) __bf16 v16bf;
typedef __attribute__((ext_vector_type(8)))  float  v8f;

#define NUM_HEADS 32
#define NUM_KV    8
#define HD        128
#define GQA       4
#define QTILE     32
#define KTILE     32
#define KSTRIDE   136   // K tile LDS row stride (bf16 elems): 272B rows, 16B aligned
#define VSTRIDE   40    // V^T tile LDS row stride (bf16 elems): 80B rows, 16B aligned
// SCALE * log2(e): fold into Q so softmax uses exp2 directly
#define SCALE_LOG2E 0.1275103276352009f

// K index inside a 16-bit A-fragment (16x32) for half-element i, per ISA 7.12.2:
// lanes 0-15 hold K in {0..7, 16..23}; lanes 16-31 hold K in {8..15, 24..31}.
__device__ __forceinline__ int a_frag_k(int half, int i) {
    int vv = i >> 1, p = i & 1;
    int kk = (vv < 4) ? (vv * 2 + p) : (16 + (vv - 4) * 2 + p);
    return kk + half * 8;
}

__device__ __forceinline__ unsigned bf_bits(float f) {
    return (unsigned)__builtin_bit_cast(unsigned short, (__bf16)f);
}

// Raw v_exp_f32: args are always <= 0 here; flush-to-zero on the -1e30 mask
// is exactly the wanted semantics, so skip libm's denormal fixup.
__device__ __forceinline__ float fast_exp2(float x) {
    return __builtin_amdgcn_exp2f(x);
}

// Wave-relative LDS byte offset of a generic __shared__ pointer
// (the low 32 bits of the generic address).
__device__ __forceinline__ unsigned lds_off32(const void* p) {
    return (unsigned)(unsigned long long)p;
}

// Async global -> LDS 16-byte copy (per-lane addresses), tracked by ASYNCcnt.
__device__ __forceinline__ void async_b128(unsigned lds_byte_off, const void* gptr) {
    asm volatile("global_load_async_to_lds_b128 %0, %1, off"
                 :: "v"(lds_byte_off), "v"((unsigned long long)gptr)
                 : "memory");
}
__device__ __forceinline__ void wait_async_le8() {
    asm volatile("s_wait_asynccnt 0x8" ::: "memory");
}
__device__ __forceinline__ void wait_async_0() {
    asm volatile("s_wait_asynccnt 0x0" ::: "memory");
}

__global__ __launch_bounds__(256)
void fa_varlen_gqa_kernel(const float* __restrict__ q,
                          const float* __restrict__ k,
                          const float* __restrict__ v,
                          const int*   __restrict__ cu,
                          float*       __restrict__ out)
{
    __shared__ float  stgK[2][KTILE * HD];     // raw fp32 K tiles (async landing zone)
    __shared__ float  stgV[2][KTILE * HD];     // raw fp32 V tiles
    __shared__ __bf16 Ks[KTILE * KSTRIDE];     // [kpos][d]   bf16 row-major
    __shared__ __bf16 Vt[HD * VSTRIDE];        // [d][kpos]   bf16 transposed
    __shared__ __bf16 Pl[8][32 * 16];          // per-wave P in A-frag linear layout

    const int b  = blockIdx.x;   // sequence
    const int hk = blockIdx.y;   // kv head
    const int qt = blockIdx.z;   // 32-row query tile
    const int s0 = cu[b];
    const int L  = cu[b + 1] - s0;
    const int q0 = qt * QTILE;
    if (q0 >= L) return;         // block-uniform exit

    const int tid  = threadIdx.x;
    const int w    = tid >> 5;   // wave 0..7
    const int lane = tid & 31;
    const int half = lane >> 4;  // 0 or 1
    const int ln   = lane & 15;

    const int h  = hk * GQA + (w & 3);    // query head for this wave
    const int m0 = q0 + (w >> 2) * 16;    // wave's 16-row q sub-tile base

    // staging slice owned by this thread: key row kp, 16-float column chunk c0
    const int kp = tid >> 3;          // 0..31
    const int c0 = (tid & 7) * 16;    // 0..112

    // ---- persistent Q A-fragments (pre-scaled by SCALE*log2e), float2 loads ----
    v16bf qf[4];
    {
        const int  qrow = m0 + ln;                 // A-row M = lane&15
        const bool qv   = (qrow < L);
        const float* qp = q + (long)(s0 + (qv ? qrow : 0)) * (NUM_HEADS * HD) + h * HD;
        for (int c = 0; c < 4; ++c)
            for (int vv = 0; vv < 8; ++vv) {
                int k0 = a_frag_k(half, vv * 2);   // even: pair (k0, k0+1) contiguous
                float2 f2 = *(const float2*)(qp + c * 32 + k0);
                qf[c][vv * 2]     = (__bf16)(qv ? f2.x * SCALE_LOG2E : 0.0f);
                qf[c][vv * 2 + 1] = (__bf16)(qv ? f2.y * SCALE_LOG2E : 0.0f);
            }
    }

    // all-ones B fragment: rowsum(P) = P x 1 via one extra WMMA
    v16bf onesB;
    for (int i = 0; i < 16; ++i) onesB[i] = (__bf16)1.0f;

    v8f   o[8] = {};             // O tile: 16 rows x 128 cols fp32
    float mrow[8], lrow[8];
    for (int r = 0; r < 8; ++r) { mrow[r] = -1e30f; lrow[r] = 0.0f; }

    const int kend = min(L, q0 + QTILE);              // causal bound for block
    const int nkt  = (kend + KTILE - 1) / KTILE;      // uniform across block

    // ---- prologue: async-issue tile 0 into buffer 0 ----
    {
        int kpos = min(kp, L - 1);                    // clamp OOB rows (masked later)
        long t = (long)(s0 + kpos) * (NUM_KV * HD) + hk * HD + c0;
        unsigned lk = lds_off32(&stgK[0][tid * 16]);
        unsigned lv = lds_off32(&stgV[0][tid * 16]);
        for (int cc = 0; cc < 4; ++cc) {
            async_b128(lk + cc * 16, k + t + cc * 4);
            async_b128(lv + cc * 16, v + t + cc * 4);
        }
    }

    for (int jt = 0; jt < nkt; ++jt) {
        const int j0  = jt * KTILE;
        const int buf = jt & 1;

        // ---- issue next tile's async copies, then wait for this tile ----
        if (jt + 1 < nkt) {
            int kpos = min(j0 + KTILE + kp, L - 1);
            long t = (long)(s0 + kpos) * (NUM_KV * HD) + hk * HD + c0;
            unsigned lk = lds_off32(&stgK[buf ^ 1][tid * 16]);
            unsigned lv = lds_off32(&stgV[buf ^ 1][tid * 16]);
            for (int cc = 0; cc < 4; ++cc) {
                async_b128(lk + cc * 16, k + t + cc * 4);
                async_b128(lv + cc * 16, v + t + cc * 4);
            }
            wait_async_le8();    // in-order returns: this tile's 8 copies landed
        } else {
            wait_async_0();
        }

        // ---- convert own slice: fp32 stage -> bf16 Ks (row-major) + Vt (transposed)
        {
            const float* sk = &stgK[buf][tid * 16];
            float4 f[4];
            for (int e = 0; e < 4; ++e) f[e] = *(const float4*)(sk + 4 * e);
            unsigned pk[8];
            for (int e = 0; e < 4; ++e) {
                pk[2 * e]     = (bf_bits(f[e].y) << 16) | bf_bits(f[e].x);
                pk[2 * e + 1] = (bf_bits(f[e].w) << 16) | bf_bits(f[e].z);
            }
            uint4 lo = {pk[0], pk[1], pk[2], pk[3]};
            uint4 hi = {pk[4], pk[5], pk[6], pk[7]};
            *(uint4*)&Ks[kp * KSTRIDE + c0]     = lo;
            *(uint4*)&Ks[kp * KSTRIDE + c0 + 8] = hi;

            const float* sv = &stgV[buf][tid * 16];
            float4 g[4];
            for (int e = 0; e < 4; ++e) g[e] = *(const float4*)(sv + 4 * e);
            for (int e = 0; e < 4; ++e) {
                Vt[(c0 + 4 * e + 0) * VSTRIDE + kp] = (__bf16)g[e].x;
                Vt[(c0 + 4 * e + 1) * VSTRIDE + kp] = (__bf16)g[e].y;
                Vt[(c0 + 4 * e + 2) * VSTRIDE + kp] = (__bf16)g[e].z;
                Vt[(c0 + 4 * e + 3) * VSTRIDE + kp] = (__bf16)g[e].w;
            }
        }
        __syncthreads();

        // ---- S = Q * K^T : two 16x16 tiles over 32 key columns ----
        v8f s[2] = {};
        for (int nt = 0; nt < 2; ++nt) {
            for (int c = 0; c < 4; ++c) {
                v16bf bf;
                const __bf16* kbase = Ks + (nt * 16 + ln) * KSTRIDE + c * 32;
                for (int i = 0; i < 16; ++i) bf[i] = kbase[a_frag_k(half, i)];
                s[nt] = __builtin_amdgcn_wmma_f32_16x16x32_bf16(
                    false, qf[c], false, bf, (short)0, s[nt], false, false);
            }
        }

        // ---- causal/ragged mask: only for diagonal / sequence-end tiles ----
        // (wave-uniform branch; no barriers inside)
        if ((j0 + KTILE > m0) || (j0 + KTILE > L)) {
            for (int nt = 0; nt < 2; ++nt) {
                int kc = j0 + nt * 16 + ln;
                for (int r = 0; r < 8; ++r) {
                    int  qr = m0 + r + half * 8;
                    bool ok = (kc <= qr) && (kc < L);
                    s[nt][r] = ok ? s[nt][r] : -1e30f;   // log2 domain
                }
            }
        }

        // ---- online softmax, base-2: shuffle-max, WMMA row-sum ----
        float alpha[8];
        for (int r = 0; r < 8; ++r) {
            float mx = fmaxf(s[0][r], s[1][r]);
            for (int m = 1; m < 16; m <<= 1)
                mx = fmaxf(mx, __shfl_xor(mx, m, 32));
            float mn = fmaxf(mrow[r], mx);
            alpha[r] = fast_exp2(mrow[r] - mn);
            s[0][r]  = fast_exp2(s[0][r] - mn);
            s[1][r]  = fast_exp2(s[1][r] - mn);
            mrow[r]  = mn;
        }

        // ---- P -> LDS directly in consumer A-fragment linear layout ----
        for (int nt = 0; nt < 2; ++nt) {
            int Kc = nt * 16 + ln;
            int hc = (Kc >> 3) & 1;
            int Kb = Kc - 8 * hc;
            int slot = (Kb & 7) + 8 * (Kb >> 4);
            for (int r = 0; r < 8; ++r) {
                int M = r + 8 * half;
                Pl[w][(M + 16 * hc) * 16 + slot] = (__bf16)s[nt][r];
            }
        }

        // ---- reload P as A-fragment: 32 contiguous bytes per lane ----
        v16bf pf;
        {
            const __bf16* pbase = Pl[w] + lane * 16;
            for (int i = 0; i < 16; ++i) pf[i] = pbase[i];
        }

        // ---- row sums via WMMA: rs = P x ones (replicated across columns) ----
        {
            v8f rz = {};
            v8f rs = __builtin_amdgcn_wmma_f32_16x16x32_bf16(
                false, pf, false, onesB, (short)0, rz, false, false);
            for (int r = 0; r < 8; ++r)
                lrow[r] = lrow[r] * alpha[r] + rs[r];
        }

        // ---- O += P * V : 8 d-chunks; V^T makes B-frags contiguous ----
        for (int dc = 0; dc < 8; ++dc) {
            v16bf vfrag;
            const __bf16* vbase = Vt + (dc * 16 + ln) * VSTRIDE;
            for (int i = 0; i < 16; ++i) vfrag[i] = vbase[a_frag_k(half, i)];
            v8f oc = o[dc];
            for (int r = 0; r < 8; ++r) oc[r] *= alpha[r];   // flash rescale
            o[dc] = __builtin_amdgcn_wmma_f32_16x16x32_bf16(
                false, pf, false, vfrag, (short)0, oc, false, false);
        }
        __syncthreads();   // protect Ks/Vt before next tile's overwrite
    }

    // ---- epilogue: normalize and store fp32 [T, H, D] ----
    // lrow >= 1 always (diagonal contributes exp2(0)=1), so raw v_rcp is safe.
    for (int r = 0; r < 8; ++r) {
        int qr = m0 + r + half * 8;
        if (qr < L) {
            float inv  = __builtin_amdgcn_rcpf(lrow[r]);
            long  base = (long)(s0 + qr) * (NUM_HEADS * HD) + h * HD;
            for (int dc = 0; dc < 8; ++dc)
                out[base + dc * 16 + ln] = o[dc][r] * inv;
        }
    }
}

extern "C" void kernel_launch(void* const* d_in, const int* in_sizes, int n_in,
                              void* d_out, int out_size, void* d_ws, size_t ws_size,
                              hipStream_t stream) {
    const float* q  = (const float*)d_in[0];
    const float* k  = (const float*)d_in[1];
    const float* v  = (const float*)d_in[2];
    const int*   cu = (const int*)d_in[3];
    float* out = (float*)d_out;

    const int T = in_sizes[0] / (NUM_HEADS * HD);   // total tokens
    const int B = in_sizes[3] - 1;                  // number of sequences
    const int maxTiles = (T + QTILE - 1) / QTILE;   // upper bound on per-seq q tiles

    dim3 grid(B, NUM_KV, maxTiles);
    dim3 block(256);
    fa_varlen_gqa_kernel<<<grid, block, 0, stream>>>(q, k, v, cu, out);
}
